// MultiheadAttention_4320737099942
// MI455X (gfx1250) — matrix operand order repaired and checked by reading, not compiled, not run on hardware
//
#include <hip/hip_runtime.h>

typedef __attribute__((ext_vector_type(16))) _Float16     v16h;
typedef __attribute__((ext_vector_type(8)))  float        v8f;
typedef __attribute__((ext_vector_type(4)))  unsigned int v4u;
typedef __attribute__((ext_vector_type(8)))  int          v8i;
typedef __attribute__((ext_vector_type(4)))  int          v4i;

#define EMBED 1024
#define NHEAD 16
#define HD    64
#define TSEQ  2048
#define NB    4
#define QTILE 16
#define SROW  (TSEQ + 8)   // f16 elems per LDS score row (+16B pad vs bank conflicts)

// ---------------------------------------------------------------------------
// Tensor Data Mover: async DMA of a 2-D fp32 tile (tile_w x tile_h elements,
// row stride row_stride elements) from global memory into LDS at lds_off.
// D# layout per CDNA5 ISA ch.8 (group0 128b, group1 256b, groups 2/3 zero).
// Tracked with TENSORcnt -> s_wait_tensorcnt.
// ---------------------------------------------------------------------------
__device__ __forceinline__ void tdm_load_2d(unsigned int lds_off,
                                            const float* gptr,
                                            unsigned int tile_w,
                                            unsigned int tile_h,
                                            unsigned int row_stride)
{
    const unsigned long long ga = (unsigned long long)(uintptr_t)gptr;

    v4u g0;
    g0[0] = 1u;                                            // count=1 (valid D#)
    g0[1] = lds_off;                                       // LDS byte address
    g0[2] = (unsigned int)(ga & 0xFFFFFFFFu);              // global_addr[31:0]
    g0[3] = (unsigned int)((ga >> 32) & 0x01FFFFFFu)       // global_addr[56:32]
          | (2u << 30);                                    // type=2 ("image")

    v8i g1;
    g1[0] = (int)(2u << 16);            // workgroup_mask=0, data_size=2 (4B)
    g1[1] = (int)(tile_w << 16);        // tensor_dim0[15:0]  (bits 63:48)
    g1[2] = (int)(tile_h << 16);        // tensor_dim1[15:0]  (bits 95:80)
    g1[3] = (int)(tile_w << 16);        // tile_dim0          (bits 127:112)
    g1[4] = (int)tile_h;                // tile_dim1          (bits 143:128)
    g1[5] = (int)row_stride;            // tensor_dim0_stride low 32b
    g1[6] = 0;                          // stride high / dim1 stride
    g1[7] = 0;

    v4i z = {0, 0, 0, 0};
#if defined(__clang_major__) && (__clang_major__ >= 23)
    v8i z8 = {0, 0, 0, 0, 0, 0, 0, 0};
    __builtin_amdgcn_tensor_load_to_lds(g0, g1, z, z, z8, 0);
#else
    __builtin_amdgcn_tensor_load_to_lds(g0, g1, z, z, 0);
#endif
}

__global__ __launch_bounds__(128, 1)
void mha_causal_wmma(const float* __restrict__ Q,
                     const float* __restrict__ K,
                     const float* __restrict__ V,
                     float* __restrict__ Yout,
                     float* __restrict__ Wout)
{
    __shared__ _Float16 Ss[QTILE][SROW];                 // 16 x 2056 f16 = 64.25 KB
    __shared__ float    red[QTILE][9];                   // softmax reduction scratch
    // Per-wave double-buffered DMA staging: phase 1 = 16x64 f32 K tile (4 KB),
    // phase 3 = 32x16 f32 V tile (2 KB). 4 waves x 2 bufs x 4 KB = 32 KB.
    __shared__ __align__(16) unsigned char stage[4][2][16 * 64 * 4];

    const int qt = blockIdx.x;   // query tile (T/16 = 128)
    const int h  = blockIdx.y;   // head
    const int nb = blockIdx.z;   // batch
    const int q0 = qt * QTILE;

    const int tid  = threadIdx.x;
    const int lane = tid & 31;
    const int wave = tid >> 5;   // 4 waves

    // A-operand lane mapping (16-bit A 16x32, ISA 7.12.2): lanes 0-15 hold row
    // M=lane with K={0..7,16..23}; lanes 16-31 hold row M=lane-16, K={8..15,24..31}.
    const int arow = lane & 15;
    const int koff = (lane >> 4) * 8;
    // B/C/D lane mapping: lane%16 = N column; (lane>=16) selects upper half.
    const int bcol  = lane & 15;
    const int brow0 = (lane >> 4) * 16;
    const int mbase = (lane >> 4) << 3;

    const unsigned int stage0 = (unsigned int)(uintptr_t)&stage[wave][0][0];
    const unsigned int stage1 = (unsigned int)(uintptr_t)&stage[wave][1][0];

    // ---------------- load Q fragments (A operand), head_dim split 2x32 ----
    const size_t qbase = ((size_t)nb * TSEQ + (size_t)(q0 + arow)) * EMBED + (size_t)h * HD;
    v16h a_lo, a_hi;
#pragma unroll
    for (int j = 0; j < 8; ++j) {
        a_lo[j]     = (_Float16)Q[qbase +      koff + j];
        a_lo[j + 8] = (_Float16)Q[qbase + 16 + koff + j];
        a_hi[j]     = (_Float16)Q[qbase + 32 + koff + j];
        a_hi[j + 8] = (_Float16)Q[qbase + 48 + koff + j];
    }

    // ---------------- phase 1: S = scale * Q K^T, causal mask -> LDS f16 ---
    // Each wave handles 32 key tiles of 16 keys; K tiles (16x64 fp32, row
    // stride 1024) arrive by TDM DMA, double buffered.
    const float scale = 0.125f;  // 1/sqrt(64)
    const int   kt0   = wave * 32;
    const size_t kheadbase = (size_t)nb * TSEQ * EMBED + (size_t)h * HD;

    tdm_load_2d(stage0, K + kheadbase + (size_t)(kt0 * 16) * EMBED, HD, 16, EMBED);

    for (int t = 0; t < 32; ++t) {
        const int kt   = kt0 + t;
        const int key0 = kt * 16;
        const int key  = key0 + bcol;

        if (t + 1 < 32) {
            tdm_load_2d((t & 1) ? stage0 : stage1,
                        K + kheadbase + (size_t)((kt + 1) * 16) * EMBED, HD, 16, EMBED);
            __builtin_amdgcn_s_wait_tensorcnt(1);   // oldest (current tile) done
        } else {
            __builtin_amdgcn_s_wait_tensorcnt(0);
        }

        const float* Kt = (const float*)&stage[wave][t & 1][0];   // [16 keys][64 d]
        const float* kp = Kt + bcol * HD + koff;                  // d = koff + {0..7,16..23}

        v16h b_lo, b_hi;
#pragma unroll
        for (int j = 0; j < 8; ++j) {
            b_lo[j]     = (_Float16)kp[j];
            b_lo[j + 8] = (_Float16)kp[16 + j];
            b_hi[j]     = (_Float16)kp[32 + j];
            b_hi[j + 8] = (_Float16)kp[48 + j];
        }

        v8f acc = {};
        acc = __builtin_amdgcn_wmma_f32_16x16x32_f16(false, a_lo, false, b_lo,
                                                     (short)0, acc, false, false);
        acc = __builtin_amdgcn_wmma_f32_16x16x32_f16(false, a_hi, false, b_hi,
                                                     (short)0, acc, false, false);

#pragma unroll
        for (int i = 0; i < 8; ++i) {
            const int m  = i + mbase;          // C/D row within tile
            const int qq = q0 + m;
            float s = acc[i] * scale;
            if (key > qq) s = -1e30f;          // causal mask (-> -inf in f16)
            Ss[m][key0 + bcol] = (_Float16)s;
        }
    }
    __syncthreads();

    // ---------------- phase 2: row softmax in LDS, write weights ----------
    const int row = tid >> 3;   // 16 rows
    const int sub = tid & 7;    // 8 threads per row

    float mx = -3.0e38f;
    for (int c = sub; c < TSEQ; c += 8) {
        mx = fmaxf(mx, (float)Ss[row][c]);
    }
    red[row][sub] = mx;
    __syncthreads();
    if (tid < QTILE) {
        float m = red[tid][0];
#pragma unroll
        for (int j = 1; j < 8; ++j) m = fmaxf(m, red[tid][j]);
        red[tid][8] = m;
    }
    __syncthreads();
    mx = red[row][8];

    float sum = 0.0f;
    for (int c = sub; c < TSEQ; c += 8) {
        sum += __expf((float)Ss[row][c] - mx);
    }
    __syncthreads();
    red[row][sub] = sum;
    __syncthreads();
    if (tid < QTILE) {
        float s = red[tid][0];
#pragma unroll
        for (int j = 1; j < 8; ++j) s += red[tid][j];
        red[tid][8] = s;
    }
    __syncthreads();
    const float inv = 1.0f / red[row][8];

    // Weights are written once and never re-read: non-temporal stores keep the
    // 1.07 GB stream from evicting the K/V reuse working set out of L2.
    float* wrow = Wout + (((size_t)nb * NHEAD + h) * TSEQ + (size_t)(q0 + row)) * TSEQ;
    for (int c = sub; c < TSEQ; c += 8) {
        const float p = __expf((float)Ss[row][c] - mx) * inv;
        __builtin_nontemporal_store(p, &wrow[c]);
        Ss[row][c] = (_Float16)p;       // f16 copy = A operand for PV
    }
    __syncthreads();

    // ---------------- phase 3: Y = P V, one 16x16 Y tile per wave ---------
    // V tiles (32 rows x 16 cols fp32, row stride 1024) staged by TDM DMA,
    // reusing this wave's staging buffers (its own TDM ops completed above).
    const size_t vheadbase = (size_t)nb * TSEQ * EMBED + (size_t)h * HD
                           + (size_t)wave * 16;

    tdm_load_2d(stage0, V + vheadbase, 16, 32, EMBED);

    v8f yacc = {};
    for (int t = 0; t < 64; ++t) {
        const int kt = t * 32;

        if (t + 1 < 64) {
            tdm_load_2d((t & 1) ? stage0 : stage1,
                        V + vheadbase + (size_t)(kt + 32) * EMBED, 16, 32, EMBED);
            __builtin_amdgcn_s_wait_tensorcnt(1);
        } else {
            __builtin_amdgcn_s_wait_tensorcnt(0);
        }

        const float* Vt = (const float*)&stage[wave][t & 1][0];   // [32 k][16 d]

        v16h pa;
#pragma unroll
        for (int j = 0; j < 8; ++j) {
            pa[j]     = Ss[arow][kt +      koff + j];
            pa[j + 8] = Ss[arow][kt + 16 + koff + j];
        }
        v16h vb;
#pragma unroll
        for (int j = 0; j < 8; ++j) {
            vb[j]     = (_Float16)Vt[(koff + j) * 16 + bcol];
            vb[j + 8] = (_Float16)Vt[(16 + koff + j) * 16 + bcol];
        }
        yacc = __builtin_amdgcn_wmma_f32_16x16x32_f16(false, pa, false, vb,
                                                      (short)0, yacc, false, false);
    }
#pragma unroll
    for (int i = 0; i < 8; ++i) {
        const int m = i + mbase;
        __builtin_nontemporal_store(
            yacc[i],
            &Yout[((size_t)nb * TSEQ + (size_t)(q0 + m)) * EMBED
                  + (size_t)h * HD + (size_t)wave * 16 + bcol]);
    }
}

extern "C" void kernel_launch(void* const* d_in, const int* in_sizes, int n_in,
                              void* d_out, int out_size, void* d_ws, size_t ws_size,
                              hipStream_t stream) {
    (void)in_sizes; (void)n_in; (void)out_size; (void)d_ws; (void)ws_size;
    const float* Q = (const float*)d_in[0];
    const float* K = (const float*)d_in[1];
    const float* V = (const float*)d_in[2];
    float* y = (float*)d_out;                                   // [4,2048,1024]
    float* w = y + (size_t)NB * TSEQ * EMBED;                   // [4,16,2048,2048]
    dim3 grid(TSEQ / QTILE, NHEAD, NB);
    mha_causal_wmma<<<grid, 128, 0, stream>>>(Q, K, V, y, w);
}
